// Attention_62216896249968
// MI455X (gfx1250) — matrix operand order repaired and checked by reading, compiled, not run on hardware
//
#include <hip/hip_runtime.h>

typedef __attribute__((ext_vector_type(16))) __bf16 v16bf;
typedef __attribute__((ext_vector_type(8)))  float  v8f;

#define N_Q   8192
#define N_KV  8192
#define DIM   128
#define BM    64          // query rows per workgroup (4 waves x 16)
#define BN    64          // keys per LDS tile
#define NT    (N_KV / BN) // 128 key tiles
#define NWAVE 4
#define KPAD  8           // row stride 272B -> conflict-free 16B-wide B reads
#define VPAD  8           // row stride 144B
#define PPAD  8

union BF16Frag { v16bf v; uint4 q[2]; };

__device__ __forceinline__ v16bf ld16(const __bf16* p) {
  BF16Frag f;
  f.q[0] = *(const uint4*)(p);
  f.q[1] = *(const uint4*)(p + 8);
  return f.v;
}
__device__ __forceinline__ v16bf ld8x2(const __bf16* p0, const __bf16* p1) {
  BF16Frag f;
  f.q[0] = *(const uint4*)(p0);
  f.q[1] = *(const uint4*)(p1);
  return f.v;
}

// Packed f32x2 -> bf16x2 (one VALU op instead of two converts)
__device__ __forceinline__ unsigned cvt_pk_bf16(float a, float b) {
  unsigned r;
  asm("v_cvt_pk_bf16_f32 %0, %1, %2" : "=v"(r) : "v"(a), "v"(b));
  return r;
}

// min-3-waves-per-EU: roomy enough for the allocator's ~270-reg software
// pipeline (no hot-loop spills, unlike a /4 cap), still bounds ballooning.
__global__ __launch_bounds__(128, 3)
void fa_fwd_bf16wmma(const float* __restrict__ Q, const float* __restrict__ K,
                     const float* __restrict__ V, float* __restrict__ O)
{
  // Double-buffered K/V^T tiles + per-wave P staging (~80 KB of the 320 KB WGP LDS)
  __shared__ __bf16 sK [2][BN][DIM + KPAD];    // K tile, row-major (bf16)
  __shared__ __bf16 sVt[2][DIM][BN + VPAD];    // V tile, transposed (bf16)
  __shared__ __bf16 sP [NWAVE][16][BN + PPAD]; // per-wave P staging

  const int tid  = threadIdx.x;
  const int wave = tid >> 5;
  const int lane = tid & 31;
  const int half = lane >> 4;   // 0: lanes 0-15, 1: lanes 16-31
  const int l16  = lane & 15;

  // Cooperative tile loader: 64x128 f32 -> bf16 (K row-major, V transposed).
  auto load_tile = [&](int ktile, int b) {
    const float* Kt = K + (size_t)ktile * BN * DIM;
    const float* Vt = V + (size_t)ktile * BN * DIM;
    #pragma unroll 4
    for (int i = 0; i < 16; ++i) {
      const int e   = tid + 128 * i;     // float4 index within tile
      const int row = e >> 5;            // key within tile
      const int col = (e & 31) * 4;      // d / dv
      const float4 kf = *(const float4*)(Kt + (size_t)row * DIM + col);
      unsigned* kp = (unsigned*)&sK[b][row][col];   // 8B-aligned (col % 4 == 0)
      kp[0] = cvt_pk_bf16(kf.x, kf.y);
      kp[1] = cvt_pk_bf16(kf.z, kf.w);
      const float4 vf = *(const float4*)(Vt + (size_t)row * DIM + col);
      sVt[b][col+0][row] = (__bf16)vf.x;
      sVt[b][col+1][row] = (__bf16)vf.y;
      sVt[b][col+2][row] = (__bf16)vf.z;
      sVt[b][col+3][row] = (__bf16)vf.w;
    }
  };

  // ---- Load Q stripe straight into 16-bit A-matrix layout (f32 -> bf16) ----
  // A 16x32 layout: lane holds row m = l16; half index h maps to
  // k = 32c + h + (h<8 ? 0 : 8) + (half ? 8 : 0)
  const size_t qrow = (size_t)blockIdx.x * BM + wave * 16 + l16;
  const float* Qrow = Q + qrow * DIM;
  v16bf Qa[4];
  #pragma unroll
  for (int c = 0; c < 4; ++c) {
    #pragma unroll
    for (int h = 0; h < 16; ++h) {
      const int d = 32*c + h + ((h < 8) ? 0 : 8) + (half ? 8 : 0);
      Qa[c][h] = (__bf16)Qrow[d];
    }
  }

  // ---- accumulators / running softmax stats (C-layout: row = r + 8*half) ----
  v8f m_run, l_run, Oacc[8];
  #pragma unroll
  for (int r = 0; r < 8; ++r) { m_run[r] = -3.0e38f; l_run[r] = 0.0f; }
  #pragma unroll
  for (int n = 0; n < 8; ++n)
    #pragma unroll
    for (int r = 0; r < 8; ++r) Oacc[n][r] = 0.0f;

  const float L2E = 1.44269504088896340736f;

  // Prologue: stage tile 0
  load_tile(0, 0);
  __syncthreads();

  for (int kt = 0; kt < NT; ++kt) {
    const int cur = kt & 1;

    // Prefetch tile kt+2 toward L2 (global_prefetch_b8)
    if (kt + 2 < NT) {
      __builtin_prefetch(K + (size_t)(kt + 2) * BN * DIM + tid * 64, 0, 0);
      __builtin_prefetch(V + (size_t)(kt + 2) * BN * DIM + tid * 64, 0, 0);
    }
    // Stage next tile into the other buffer; other waves' WMMA covers this latency.
    if (kt + 1 < NT) load_tile(kt + 1, cur ^ 1);

    // ---- S = Q K^T : 4 subtiles of 16 keys, 4 K-chunks of 32 each ----
    v8f S[4];
    #pragma unroll
    for (int t = 0; t < 4; ++t) {
      v8f acc;
      #pragma unroll
      for (int r = 0; r < 8; ++r) acc[r] = 0.0f;
      #pragma unroll
      for (int c = 0; c < 4; ++c) {
        // B 32x16 layout: lane holds col n = l16, k = 32c + h + (h<8 ? 0 : 8) + 8*half
        const __bf16* kp = &sK[cur][t*16 + l16][32*c + 8*half];
        const v16bf kb = ld8x2(kp, kp + 16);
        acc = __builtin_amdgcn_wmma_f32_16x16x32_bf16(
                  false, Qa[c], false, kb, (short)0, acc, false, false);
      }
      S[t] = acc;
    }

    // ---- online softmax: row max over tile (intra-half shuffle reduce) ----
    v8f mt;
    #pragma unroll
    for (int r = 0; r < 8; ++r)
      mt[r] = fmaxf(fmaxf(S[0][r], S[1][r]), fmaxf(S[2][r], S[3][r]));
    #pragma unroll
    for (int off = 8; off >= 1; off >>= 1)
      #pragma unroll
      for (int r = 0; r < 8; ++r)
        mt[r] = fmaxf(mt[r], __shfl_xor(mt[r], off, 32));

    v8f m_new, scl, rsum;
    #pragma unroll
    for (int r = 0; r < 8; ++r) {
      m_new[r] = fmaxf(m_run[r], mt[r]);
      scl[r]   = __builtin_amdgcn_exp2f((m_run[r] - m_new[r]) * L2E);
      rsum[r]  = 0.0f;
    }
    #pragma unroll
    for (int t = 0; t < 4; ++t)
      #pragma unroll
      for (int r = 0; r < 8; ++r) {
        const float p = __builtin_amdgcn_exp2f((S[t][r] - m_new[r]) * L2E);
        S[t][r] = p;
        rsum[r] += p;
      }
    #pragma unroll
    for (int off = 8; off >= 1; off >>= 1)
      #pragma unroll
      for (int r = 0; r < 8; ++r)
        rsum[r] += __shfl_xor(rsum[r], off, 32);
    #pragma unroll
    for (int r = 0; r < 8; ++r) {
      l_run[r] = l_run[r] * scl[r] + rsum[r];
      m_run[r] = m_new[r];
    }
    #pragma unroll
    for (int n = 0; n < 8; ++n)
      #pragma unroll
      for (int r = 0; r < 8; ++r)
        Oacc[n][r] *= scl[r];

    // ---- C-layout P -> wave-private LDS -> A-layout for PV GEMM ----
    #pragma unroll
    for (int t = 0; t < 4; ++t)
      #pragma unroll
      for (int r = 0; r < 8; ++r)
        sP[wave][r + 8*half][t*16 + l16] = (__bf16)S[t][r];

    asm volatile("s_wait_dscnt 0" ::: "memory");  // wave-private; DS in-order

    v16bf Pa[2];
    #pragma unroll
    for (int kc = 0; kc < 2; ++kc) {
      const __bf16* pr = &sP[wave][l16][32*kc + 8*half];
      Pa[kc] = ld8x2(pr, pr + 16);
    }

    // ---- O += P V : 8 dv-tiles, 2 key-chunks of 32 ----
    #pragma unroll
    for (int n = 0; n < 8; ++n) {
      #pragma unroll
      for (int kc = 0; kc < 2; ++kc) {
        const __bf16* vp = &sVt[cur][n*16 + l16][32*kc + 8*half];
        const v16bf vb = ld8x2(vp, vp + 16);
        Oacc[n] = __builtin_amdgcn_wmma_f32_16x16x32_bf16(
                      false, Pa[kc], false, vb, (short)0, Oacc[n], false, false);
      }
    }

    // Single barrier per iteration: all waves done reading buf[cur] and
    // done writing buf[cur^1] before the roles swap.
    __syncthreads();
  }

  // ---- epilogue: O / l ----
  v8f inv;
  #pragma unroll
  for (int r = 0; r < 8; ++r) inv[r] = 1.0f / l_run[r];
  const size_t obase = (size_t)blockIdx.x * BM + wave * 16;
  #pragma unroll
  for (int n = 0; n < 8; ++n)
    #pragma unroll
    for (int r = 0; r < 8; ++r) {
      const size_t row = obase + r + 8*half;
      O[row * DIM + n*16 + l16] = Oacc[n][r] * inv[r];
    }
}

extern "C" void kernel_launch(void* const* d_in, const int* in_sizes, int n_in,
                              void* d_out, int out_size, void* d_ws, size_t ws_size,
                              hipStream_t stream) {
  (void)in_sizes; (void)n_in; (void)d_ws; (void)ws_size; (void)out_size;
  const float* Q = (const float*)d_in[0];
  const float* K = (const float*)d_in[1];
  const float* V = (const float*)d_in[2];
  float* O = (float*)d_out;
  dim3 grid(N_Q / BM);
  dim3 block(128);
  hipLaunchKernelGGL(fa_fwd_bf16wmma, grid, block, 0, stream, Q, K, V, O);
}
